// multiAttentionHead_85066122265006
// MI455X (gfx1250) — matrix operand order repaired and checked by reading, not compiled, not run on hardware
//
#include <hip/hip_runtime.h>

// ---------------------------------------------------------------------------
// MHA forward, B=2 S=2048 F=1024 H=16 D=64, gfx1250 (CDNA5, wave32).
// bf16 WMMA, f32 accumulation, flash-attention fusion.
// Double-buffered async global->LDS staging (ASYNCcnt) overlaps HBM/L2
// latency with WMMA compute; one barrier per tile iteration.
// ---------------------------------------------------------------------------

typedef unsigned short u16;
typedef unsigned long long u64;
typedef __attribute__((ext_vector_type(16))) __bf16 bf16x16;
typedef __attribute__((ext_vector_type(8)))  float  f32x8;

union BFrag {
  bf16x16 v;
  uint4   q[2];  // 32 bytes = 16 bf16
};

static constexpr int Bb = 2, Ss = 2048, Ff = 1024, Hh = 16, Dd = 64;
static constexpr int Mg = Bb * Ss;   // 4096 rows for all GEMMs
static constexpr int Ng = Hh * Dd;   // 1024 cols (== F for final projection)
static constexpr int Kg = Ff;        // 1024 reduction dim

__device__ __forceinline__ u16 f32_to_bf16(float f) {
  unsigned u = __float_as_uint(f);
  u += 0x7FFFu + ((u >> 16) & 1u);   // round-to-nearest-even
  return (u16)(u >> 16);
}

__device__ __forceinline__ f32x8 wmma_bf16(const bf16x16& a, const bf16x16& b,
                                           const f32x8& c) {
  return __builtin_amdgcn_wmma_f32_16x16x32_bf16(
      false, a, false, b, (short)0, c, false, false);
}

// Async copy 16B global -> LDS (GLOBAL_LOAD_ASYNC_TO_LDS_B128, GVS mode).
__device__ __forceinline__ void async_ld_b128(unsigned lds_addr, unsigned voff,
                                              u64 saddr) {
  asm volatile("global_load_async_to_lds_b128 %0, %1, %2"
               :: "v"(lds_addr), "v"(voff), "s"(saddr) : "memory");
}
__device__ __forceinline__ void wait_async0() {
  asm volatile("s_wait_asynccnt 0" ::: "memory");
}
__device__ __forceinline__ unsigned lds_off(const void* p) {
  return (unsigned)(uintptr_t)p;  // low 32 bits of flat LDS addr = LDS offset
}

// ---------------------------------------------------------------------------
// f32 -> bf16 conversion prepass
// ---------------------------------------------------------------------------
__global__ void cvt_f32_to_bf16(const float* __restrict__ s,
                                u16* __restrict__ d, int n4) {
  int i = blockIdx.x * blockDim.x + threadIdx.x;
  if (i >= n4) return;
  const float4 f = ((const float4*)s)[i];
  uint2 o;
  o.x = (unsigned)f32_to_bf16(f.x) | ((unsigned)f32_to_bf16(f.y) << 16);
  o.y = (unsigned)f32_to_bf16(f.z) | ((unsigned)f32_to_bf16(f.w) << 16);
  ((uint2*)d)[i] = o;
}

// ---------------------------------------------------------------------------
// bf16 GEMM: C[4096,1024] = A[4096,1024] * B[1024,1024], f32 accumulate.
// Block tile 128x64, 8 waves x 32x32. K-step 32, double-buffered:
// A staged [m][k] via async-to-LDS, B staged transposed [n][k] (packed b32).
// ---------------------------------------------------------------------------
template <bool OUT_BF16>
__global__ void __launch_bounds__(256)
gemm_bf16_k(const u16* __restrict__ A, const u16* __restrict__ Bm,
            void* __restrict__ Cv, float scale) {
  __shared__ __align__(16) u16 As[2][128][32];  // 16 KB
  __shared__ __align__(16) u16 Bt[2][64][32];   // 8 KB (transposed)

  const int tid  = threadIdx.x;
  const int lane = tid & 31;
  const int wave = tid >> 5;
  const int wm   = wave & 3;
  const int wn   = wave >> 2;
  const int tcol = lane & 15;
  const int hf8  = (lane >> 4) * 8;
  const int hf16 = (lane >> 4) * 16;
  const int gm0  = blockIdx.y * 128;
  const int gn0  = blockIdx.x * 64;

  f32x8 acc[2][2];
#pragma unroll
  for (int i = 0; i < 2; ++i)
#pragma unroll
    for (int j = 0; j < 2; ++j)
#pragma unroll
      for (int r = 0; r < 8; ++r) acc[i][j][r] = 0.0f;

  const int am0 = tid >> 2, ac = (tid & 3) * 8;  // A staging: 2 tasks/thread
  const int bkp = tid >> 3, bc = (tid & 7) * 8;  // B staging: tid<128 only

  auto stage = [&](int bi, int k0) {
    const u64 abase = (u64)(uintptr_t)A + 2ull * ((size_t)gm0 * Kg + k0);
#pragma unroll
    for (int i = 0; i < 2; ++i) {
      int mm = am0 + i * 64;
      async_ld_b128(lds_off(&As[bi][mm][ac]),
                    (unsigned)(2 * (mm * Kg + ac)), abase);
    }
    if (tid < 128) {
      const u16* bsrc = &Bm[(size_t)(k0 + 2 * bkp) * Ng + gn0 + bc];
      uint4 b0 = *(const uint4*)bsrc;
      uint4 b1 = *(const uint4*)(bsrc + Ng);
      const u16* e0 = (const u16*)&b0;
      const u16* e1 = (const u16*)&b1;
#pragma unroll
      for (int i = 0; i < 8; ++i)
        *(unsigned*)&Bt[bi][bc + i][2 * bkp] =
            (unsigned)e0[i] | ((unsigned)e1[i] << 16);
    }
  };

  stage(0, 0);
  wait_async0();
  __syncthreads();

  constexpr int NK = Kg / 32;  // 32
#pragma unroll 2
  for (int kt = 0; kt < NK; ++kt) {
    const int cur = kt & 1;
    if (kt + 1 < NK) stage(cur ^ 1, (kt + 1) * 32);  // overlap next tile

    const u16(*as)[32] = As[cur];
    const u16(*bt)[32] = Bt[cur];
    BFrag a[2], b[2];
#pragma unroll
    for (int mi = 0; mi < 2; ++mi) {
      int row = wm * 32 + mi * 16 + tcol;
      a[mi].q[0] = *(const uint4*)&as[row][hf8];
      a[mi].q[1] = *(const uint4*)&as[row][hf8 + 16];
    }
#pragma unroll
    for (int ni = 0; ni < 2; ++ni) {
      int col = wn * 32 + ni * 16 + tcol;
      b[ni].q[0] = *(const uint4*)&bt[col][hf8];
      b[ni].q[1] = *(const uint4*)&bt[col][hf8 + 16];
    }
#pragma unroll
    for (int mi = 0; mi < 2; ++mi)
#pragma unroll
      for (int ni = 0; ni < 2; ++ni)
        acc[mi][ni] = wmma_bf16(a[mi].v, b[ni].v, acc[mi][ni]);

    wait_async0();   // next-tile async ops had the whole compute to finish
    __syncthreads();
  }

#pragma unroll
  for (int mi = 0; mi < 2; ++mi)
#pragma unroll
    for (int ni = 0; ni < 2; ++ni) {
      int mrow0 = gm0 + wm * 32 + mi * 16 + hf8;
      int col   = gn0 + wn * 32 + ni * 16 + tcol;
#pragma unroll
      for (int r = 0; r < 8; ++r) {
        float val = acc[mi][ni][r] * scale;
        size_t off = (size_t)(mrow0 + r) * Ng + col;
        if (OUT_BF16)
          ((u16*)Cv)[off] = f32_to_bf16(val);
        else
          ((float*)Cv)[off] = val;
      }
    }
}

// ---------------------------------------------------------------------------
// Flash attention: block = (b, h, 128 q rows), 8 waves x 16 rows.
// 64-key tiles, double-buffered: K staged [t][d] via async-to-LDS, V staged
// transposed [d][t] (packed b32). Row max via shfl butterfly in 16-lane
// halves (matches WMMA C layout); row sums via WMMA vs all-ones B fragment.
// q pre-scaled by 1/sqrt(D).
// ---------------------------------------------------------------------------
__global__ void __launch_bounds__(256)
attn_k(const u16* __restrict__ qg, const u16* __restrict__ kg,
       const u16* __restrict__ vg, u16* __restrict__ ctx) {
  __shared__ __align__(16) u16 Ks[2][64][64];   // 16 KB [t][d]
  __shared__ __align__(16) u16 Vt[2][64][64];   // 16 KB [d][t]
  __shared__ __align__(16) u16 Ps[8][16][64];   // 16 KB per-wave P staging

  const int tid  = threadIdx.x;
  const int lane = tid & 31;
  const int wave = tid >> 5;
  const int tcol = lane & 15;
  const int hf   = lane >> 4;
  const int b = blockIdx.z, h = blockIdx.y;
  const int q0 = blockIdx.x * 128 + wave * 16;

  // Q A-fragments: 16 rows x 64 d = 2 K-steps
  const u16* qrow = &qg[(((size_t)b * Ss + q0 + tcol) * Hh + h) * Dd];
  BFrag qa0, qa1;
  qa0.q[0] = *(const uint4*)&qrow[hf * 8];
  qa0.q[1] = *(const uint4*)&qrow[hf * 8 + 16];
  qa1.q[0] = *(const uint4*)&qrow[32 + hf * 8];
  qa1.q[1] = *(const uint4*)&qrow[32 + hf * 8 + 16];

  BFrag ones;  // bf16 1.0 everywhere, for the row-sum WMMA
#pragma unroll
  for (int i = 0; i < 16; ++i) ((u16*)&ones)[i] = 0x3f80;

  float m[8], l[8];
  f32x8 O[4];
#pragma unroll
  for (int r = 0; r < 8; ++r) { m[r] = -3.0e38f; l[r] = 0.0f; }
#pragma unroll
  for (int n = 0; n < 4; ++n)
#pragma unroll
    for (int r = 0; r < 8; ++r) O[n][r] = 0.0f;

  const int krow0 = tid >> 3, kc = (tid & 7) * 8;  // K: rows +0,+32
  const int vrp = tid >> 3, vc = (tid & 7) * 8;    // V: row pair 2vrp,2vrp+1

  auto stage = [&](int bi, int t0) {
    const size_t tilebase = (((size_t)b * Ss + t0) * Hh + h) * Dd;
    const u64 kbase = (u64)(uintptr_t)kg + 2ull * tilebase;
#pragma unroll
    for (int i = 0; i < 2; ++i) {
      int row = krow0 + i * 32;
      async_ld_b128(lds_off(&Ks[bi][row][kc]),
                    (unsigned)(2 * (row * Hh * Dd + kc)), kbase);
    }
    const u16* vsrc = &vg[tilebase + (size_t)(2 * vrp) * Hh * Dd + vc];
    uint4 a0 = *(const uint4*)vsrc;
    uint4 a1 = *(const uint4*)(vsrc + Hh * Dd);
    const u16* e0 = (const u16*)&a0;
    const u16* e1 = (const u16*)&a1;
#pragma unroll
    for (int i = 0; i < 8; ++i)
      *(unsigned*)&Vt[bi][vc + i][2 * vrp] =
          (unsigned)e0[i] | ((unsigned)e1[i] << 16);
  };

  stage(0, 0);
  wait_async0();
  __syncthreads();

  constexpr int NT = Ss / 64;  // 32
#pragma unroll 2
  for (int it = 0; it < NT; ++it) {
    const int cur = it & 1;
    if (it + 1 < NT) stage(cur ^ 1, (it + 1) * 64);  // overlap next tile

    const u16(*ks)[64] = Ks[cur];
    const u16(*vt)[64] = Vt[cur];

    // --- scores: 16 rows x 64 keys (4 col tiles x 2 K-steps) ---
    f32x8 s[4];
#pragma unroll
    for (int ct = 0; ct < 4; ++ct) {
      BFrag kf;
      kf.q[0] = *(const uint4*)&ks[ct * 16 + tcol][hf * 8];
      kf.q[1] = *(const uint4*)&ks[ct * 16 + tcol][hf * 8 + 16];
      f32x8 z;
#pragma unroll
      for (int r = 0; r < 8; ++r) z[r] = 0.0f;
      s[ct] = wmma_bf16(qa0.v, kf.v, z);
      kf.q[0] = *(const uint4*)&ks[ct * 16 + tcol][32 + hf * 8];
      kf.q[1] = *(const uint4*)&ks[ct * 16 + tcol][32 + hf * 8 + 16];
      s[ct] = wmma_bf16(qa1.v, kf.v, s[ct]);
    }

    // --- online softmax: row max (butterfly over 16-lane half) ---
    float mt[8];
#pragma unroll
    for (int r = 0; r < 8; ++r)
      mt[r] = fmaxf(fmaxf(s[0][r], s[1][r]), fmaxf(s[2][r], s[3][r]));
#pragma unroll
    for (int off = 1; off < 16; off <<= 1)
#pragma unroll
      for (int r = 0; r < 8; ++r)
        mt[r] = fmaxf(mt[r], __shfl_xor(mt[r], off, 32));

    float p[4][8];
#pragma unroll
    for (int r = 0; r < 8; ++r) {
      float mn    = fmaxf(m[r], mt[r]);
      float alpha = __expf(m[r] - mn);
      m[r] = mn;
      l[r] *= alpha;
#pragma unroll
      for (int ct = 0; ct < 4; ++ct) p[ct][r] = __expf(s[ct][r] - mn);
#pragma unroll
      for (int n = 0; n < 4; ++n) O[n][r] *= alpha;
    }

    // --- P: C-layout -> A-layout via per-wave LDS bounce (bf16) ---
#pragma unroll
    for (int ct = 0; ct < 4; ++ct)
#pragma unroll
      for (int r = 0; r < 8; ++r)
        Ps[wave][hf * 8 + r][ct * 16 + tcol] = f32_to_bf16(p[ct][r]);
    BFrag pa0, pa1;  // same-wave LDS ops are in order
    pa0.q[0] = *(const uint4*)&Ps[wave][tcol][hf * 8];
    pa0.q[1] = *(const uint4*)&Ps[wave][tcol][hf * 8 + 16];
    pa1.q[0] = *(const uint4*)&Ps[wave][tcol][32 + hf * 8];
    pa1.q[1] = *(const uint4*)&Ps[wave][tcol][32 + hf * 8 + 16];

    // --- row sums via WMMA with all-ones B (lands in matching C layout) ---
    f32x8 rsum;
#pragma unroll
    for (int r = 0; r < 8; ++r) rsum[r] = 0.0f;
    rsum = wmma_bf16(pa0.v, ones.v, rsum);
    rsum = wmma_bf16(pa1.v, ones.v, rsum);
#pragma unroll
    for (int r = 0; r < 8; ++r) l[r] += rsum[r];

    // --- O += P * V  (2 K-steps of 32 keys, 4 d-column tiles) ---
#pragma unroll
    for (int n = 0; n < 4; ++n) {
      BFrag vb;
      vb.q[0] = *(const uint4*)&vt[n * 16 + tcol][hf * 8];
      vb.q[1] = *(const uint4*)&vt[n * 16 + tcol][hf * 8 + 16];
      O[n] = wmma_bf16(pa0.v, vb.v, O[n]);
      vb.q[0] = *(const uint4*)&vt[n * 16 + tcol][32 + hf * 8];
      vb.q[1] = *(const uint4*)&vt[n * 16 + tcol][32 + hf * 8 + 16];
      O[n] = wmma_bf16(pa1.v, vb.v, O[n]);
    }

    wait_async0();
    __syncthreads();
  }

  // normalize and store ctx (bf16) [B,S,H,D]
  float inv[8];
#pragma unroll
  for (int r = 0; r < 8; ++r) inv[r] = 1.0f / l[r];
#pragma unroll
  for (int n = 0; n < 4; ++n) {
    int d = n * 16 + tcol;
#pragma unroll
    for (int r = 0; r < 8; ++r) {
      int srow = q0 + hf * 8 + r;
      ctx[(((size_t)b * Ss + srow) * Hh + h) * Dd + d] =
          f32_to_bf16(O[n][r] * inv[r]);
    }
  }
}

// ---------------------------------------------------------------------------
// Launcher. Workspace: xb/ctx 8MB @0, weights 2MB each @8..16MB,
// q/k/v 8MB each @16/24/32MB. Total 40MB.
// ---------------------------------------------------------------------------
extern "C" void kernel_launch(void* const* d_in, const int* in_sizes, int n_in,
                              void* d_out, int out_size, void* d_ws,
                              size_t ws_size, hipStream_t stream) {
  (void)in_sizes; (void)n_in; (void)out_size; (void)ws_size;
  const float* x  = (const float*)d_in[0];
  const float* Wq = (const float*)d_in[1];
  const float* Wk = (const float*)d_in[2];
  const float* Wv = (const float*)d_in[3];
  const float* Wo = (const float*)d_in[4];

  char* ws = (char*)d_ws;
  u16* xb  = (u16*)(ws);
  u16* wqb = (u16*)(ws + ((size_t)8  << 20));
  u16* wkb = (u16*)(ws + ((size_t)10 << 20));
  u16* wvb = (u16*)(ws + ((size_t)12 << 20));
  u16* wob = (u16*)(ws + ((size_t)14 << 20));
  u16* qb  = (u16*)(ws + ((size_t)16 << 20));
  u16* kb  = (u16*)(ws + ((size_t)24 << 20));
  u16* vb  = (u16*)(ws + ((size_t)32 << 20));
  u16* ctx = xb;  // x dead after QKV projections

  const int NX = Bb * Ss * Ff;
  const int NW = Ff * Hh * Dd;
  cvt_f32_to_bf16<<<(NX / 4 + 255) / 256, 256, 0, stream>>>(x,  xb,  NX / 4);
  cvt_f32_to_bf16<<<(NW / 4 + 255) / 256, 256, 0, stream>>>(Wq, wqb, NW / 4);
  cvt_f32_to_bf16<<<(NW / 4 + 255) / 256, 256, 0, stream>>>(Wk, wkb, NW / 4);
  cvt_f32_to_bf16<<<(NW / 4 + 255) / 256, 256, 0, stream>>>(Wv, wvb, NW / 4);
  cvt_f32_to_bf16<<<(NW / 4 + 255) / 256, 256, 0, stream>>>(Wo, wob, NW / 4);

  dim3 gg(Ng / 64, Mg / 128);  // (16, 32)
  gemm_bf16_k<true><<<gg, 256, 0, stream>>>(xb, wqb, qb, 0.125f);  // 1/sqrt(D)
  gemm_bf16_k<true><<<gg, 256, 0, stream>>>(xb, wkb, kb, 1.0f);
  gemm_bf16_k<true><<<gg, 256, 0, stream>>>(xb, wvb, vb, 1.0f);

  attn_k<<<dim3(Ss / 128, Hh, Bb), 256, 0, stream>>>(qb, kb, vb, ctx);

  gemm_bf16_k<false><<<gg, 256, 0, stream>>>(ctx, wob, d_out, 1.0f);
}
